// GroupedExperts_73607149519033
// MI455X (gfx1250) — matrix operand order repaired and checked by reading, not compiled, not run on hardware
//
#include <hip/hip_runtime.h>
#include <hip/hip_bf16.h>

// -------- problem constants (match reference) --------
#define DIM_IN      2048
#define DIM_HID     1024
#define NUM_EXPERTS 8
#define TOKENS      4096
#define CAP         768      // (TOKENS/NUM_EXPERTS)*3/2 per-expert capacity
#define EPS         1e-6f

// -------- tiling --------
#define TM    128            // tokens per workgroup tile
#define TN    64             // output columns per workgroup tile
#define TK    64             // K depth per LDS tile (2 wmma k-steps)
#define LDSB  72             // LDS row stride in bf16 elems (64 + 8 pad = 144B, 16B aligned)
#define MTILES ((CAP + TM - 1) / TM)   // 6

typedef __attribute__((ext_vector_type(16))) __bf16 v16bf;
typedef __attribute__((ext_vector_type(8)))  float  v8f;
typedef __attribute__((ext_vector_type(4)))  int    v4i;
typedef __attribute__((address_space(1))) v4i g_v4i;   // global int4
typedef __attribute__((address_space(3))) v4i l_v4i;   // LDS int4

#if __has_builtin(__builtin_amdgcn_global_load_async_to_lds_b128)
#define HAVE_ASYNC_LDS 1
#else
#define HAVE_ASYNC_LDS 0
#endif

union FragU { v16bf v; uint4 q[2]; };

__device__ __forceinline__ unsigned short f2bf(float f) {
  unsigned u = __float_as_uint(f);
  u += 0x7FFFu + ((u >> 16) & 1u);           // round-to-nearest-even
  return (unsigned short)(u >> 16);
}
__device__ __forceinline__ float bf2f(unsigned short h) {
  return __uint_as_float(((unsigned)h) << 16);
}
// Pack 2 f32 -> packed bf16x2 dword, preferring v_cvt_pk_bf16_f32.
__device__ __forceinline__ unsigned pk2bf(float a, float b) {
#if __has_builtin(__builtin_amdgcn_cvt_pk_bf16_f32)
  return __builtin_bit_cast(unsigned, __builtin_amdgcn_cvt_pk_bf16_f32(a, b));
#else
  return (unsigned)f2bf(a) | ((unsigned)f2bf(b) << 16);
#endif
}
__device__ __forceinline__ uint2 pack4bf(float a, float b, float c, float d) {
  uint2 r;
  r.x = pk2bf(a, b);
  r.y = pk2bf(c, d);
  return r;
}

__device__ __forceinline__ void wait_async0() {
#if HAVE_ASYNC_LDS
#if __has_builtin(__builtin_amdgcn_s_wait_asynccnt)
  __builtin_amdgcn_s_wait_asynccnt(0);
#else
  asm volatile("s_wait_asynccnt 0x0" ::: "memory");
#endif
#endif
}

// A fragment: 16x32 bf16. lane&15 = M row; lane>>4 selects K halves
// {0..7,16..23} vs {8..15,24..31}; pairs packed -> two 16B LDS reads.
__device__ __forceinline__ v16bf load_fragA(const __bf16* rowbase, int sel) {
  FragU f;
  f.q[0] = *(const uint4*)(rowbase + sel * 8);
  f.q[1] = *(const uint4*)(rowbase + 16 + sel * 8);
  return f.v;
}
// B fragment: 32x16 bf16 (K x N) stored as [n][k] rows in LDS. lane&15 = N;
// lanes 0-15 hold K={0..7,16..23}, lanes 16-31 hold K={8..15,24..31}.
__device__ __forceinline__ v16bf load_fragB(const __bf16* rowbase, int sel) {
  FragU f;
  f.q[0] = *(const uint4*)(rowbase + sel * 8);
  f.q[1] = *(const uint4*)(rowbase + 16 + sel * 8);
  return f.v;
}

__device__ __forceinline__ int expert_start(const int* __restrict__ counts, int e) {
  int s = 0;
  for (int i = 0; i < e; ++i) s += counts[i];
  return s;
}

// ============================================================================
// Kernel A: h = silu(X@W1^T) * (X@W3^T) * mid_w   -> hbuf (bf16 [T, DIM_HID])
// grid: (MTILES, DIM_HID/TN, NUM_EXPERTS), block 256 (8 waves)
// ============================================================================
__global__ __launch_bounds__(256)
void moe_up_kernel(const float* __restrict__ x,
                   const float* __restrict__ w1,
                   const float* __restrict__ w3,
                   const float* __restrict__ midw,
                   const int*   __restrict__ counts,
                   unsigned short* __restrict__ hbuf) {
  __shared__ __align__(16) __bf16 sA [TM * LDSB];
  __shared__ __align__(16) __bf16 sB1[TN * LDSB];
  __shared__ __align__(16) __bf16 sB3[TN * LDSB];

  const int e     = blockIdx.z;
  const int start = expert_start(counts, e);
  const int cnt   = counts[e];
  if ((int)blockIdx.x * TM >= cnt) return;     // uniform early-out
  const int m0     = start + blockIdx.x * TM;
  const int rowEnd = start + cnt;
  const int n0     = blockIdx.y * TN;

  const int tid  = threadIdx.x;
  const int lane = tid & 31;
  const int wid  = tid >> 5;
  const int wm   = wid >> 1;       // 0..3 : 32-row strip
  const int wn   = wid & 1;        // 0..1 : 32-col strip
  const int sel  = lane >> 4;
  const int l15  = lane & 15;

  v8f acc1[2][2] = {};
  v8f acc3[2][2] = {};

  const float* w1b = w1 + ((size_t)e * DIM_HID + n0) * DIM_IN;
  const float* w3b = w3 + ((size_t)e * DIM_HID + n0) * DIM_IN;

  for (int k0 = 0; k0 < DIM_IN; k0 += TK) {
    // ---- fill A tile: 128 x 64 f32 -> bf16 (8 passes x 256 thr x float4)
    #pragma unroll
    for (int p = 0; p < 8; ++p) {
      int idx = p * 256 + tid;
      int r = idx >> 4;                 // 16 float4 groups per 64-col row
      int c = (idx & 15) * 4;
      float4 v = make_float4(0.f, 0.f, 0.f, 0.f);
      if (m0 + r < rowEnd)
        v = *(const float4*)(x + (size_t)(m0 + r) * DIM_IN + k0 + c);
      *(uint2*)&sA[r * LDSB + c] = pack4bf(v.x, v.y, v.z, v.w);
    }
    // ---- fill B tiles (w1, w3): 64 x 64 f32 each (4 passes)
    #pragma unroll
    for (int p = 0; p < 4; ++p) {
      int idx = p * 256 + tid;
      int r = idx >> 4;
      int c = (idx & 15) * 4;
      float4 a = *(const float4*)(w1b + (size_t)r * DIM_IN + k0 + c);
      float4 b = *(const float4*)(w3b + (size_t)r * DIM_IN + k0 + c);
      *(uint2*)&sB1[r * LDSB + c] = pack4bf(a.x, a.y, a.z, a.w);
      *(uint2*)&sB3[r * LDSB + c] = pack4bf(b.x, b.y, b.z, b.w);
    }
    __syncthreads();

    #pragma unroll
    for (int ks = 0; ks < 2; ++ks) {            // two 16x16x32 k-steps
      const int ko = ks * 32;
      v16bf a[2], b1f[2], b3f[2];
      #pragma unroll
      for (int i = 0; i < 2; ++i)
        a[i] = load_fragA(&sA[(wm * 32 + i * 16 + l15) * LDSB + ko], sel);
      #pragma unroll
      for (int j = 0; j < 2; ++j) {
        b1f[j] = load_fragB(&sB1[(wn * 32 + j * 16 + l15) * LDSB + ko], sel);
        b3f[j] = load_fragB(&sB3[(wn * 32 + j * 16 + l15) * LDSB + ko], sel);
      }
      #pragma unroll
      for (int i = 0; i < 2; ++i)
        #pragma unroll
        for (int j = 0; j < 2; ++j) {
          acc1[i][j] = __builtin_amdgcn_wmma_f32_16x16x32_bf16(
              false, a[i], false, b1f[j], (short)0, acc1[i][j], false, false);
          acc3[i][j] = __builtin_amdgcn_wmma_f32_16x16x32_bf16(
              false, a[i], false, b3f[j], (short)0, acc3[i][j], false, false);
        }
    }
    __syncthreads();
  }

  // ---- epilogue: SwiGLU + mid_w, store bf16
  #pragma unroll
  for (int i = 0; i < 2; ++i) {
    #pragma unroll
    for (int j = 0; j < 2; ++j) {
      const int n  = n0 + wn * 32 + j * 16 + l15;
      const float mw = midw[n];
      #pragma unroll
      for (int v = 0; v < 8; ++v) {
        const int m    = wm * 32 + i * 16 + v + sel * 8;  // C/D layout: M=v(+8)
        const int grow = m0 + m;
        if (grow < rowEnd) {
          float g1 = acc1[i][j][v];
          float g3 = acc3[i][j][v];
          float h  = (g1 / (1.0f + __expf(-g1))) * g3 * mw;
          hbuf[(size_t)grow * DIM_HID + n] = f2bf(h);
        }
      }
    }
  }
}

// ============================================================================
// Kernel B: per-token RMSNorm scale = rsqrt(mean(h^2) + eps). Wave per token.
// grid: TOKENS/8 blocks of 256.
// ============================================================================
__global__ __launch_bounds__(256)
void rms_scale_kernel(const unsigned short* __restrict__ hbuf,
                      float* __restrict__ scale) {
  const int token = blockIdx.x * 8 + (threadIdx.x >> 5);
  const int lane  = threadIdx.x & 31;
  const uint4* row = (const uint4*)(hbuf + (size_t)token * DIM_HID);
  float s = 0.f;
  #pragma unroll
  for (int i = 0; i < 4; ++i) {                // 1024 bf16 / 32 lanes / 8 per uint4
    uint4 q = row[lane + 32 * i];
    const unsigned uw[4] = {q.x, q.y, q.z, q.w};
    #pragma unroll
    for (int j = 0; j < 4; ++j) {
      float lo = bf2f((unsigned short)(uw[j] & 0xFFFFu));
      float hi = bf2f((unsigned short)(uw[j] >> 16));
      s += lo * lo + hi * hi;
    }
  }
  #pragma unroll
  for (int off = 16; off > 0; off >>= 1)
    s += __shfl_xor(s, off, 32);
  if (lane == 0)
    scale[token] = rsqrtf(s / (float)DIM_HID + EPS);
}

// ============================================================================
// Kernel C: out = scale_row * (h @ W2^T)     (f32 out [T, DIM_IN])
// Per-row RMS scale commutes out of the K-sum, so it is applied on the f32
// accumulators in the epilogue. The A tile is therefore a raw bf16 copy:
// fill it with GLOBAL_LOAD_ASYNC_TO_LDS_B128 (ASYNCcnt) when available.
// grid: (MTILES, DIM_IN/TN, NUM_EXPERTS), block 256 (8 waves)
// ============================================================================
__global__ __launch_bounds__(256)
void moe_down_kernel(const unsigned short* __restrict__ hbuf,
                     const float* __restrict__ scale,
                     const float* __restrict__ w2,
                     const int*   __restrict__ counts,
                     float* __restrict__ out) {
  __shared__ __align__(16) __bf16 sA[TM * LDSB];
  __shared__ __align__(16) __bf16 sB[TN * LDSB];

  const int e     = blockIdx.z;
  const int start = expert_start(counts, e);
  const int cnt   = counts[e];
  if ((int)blockIdx.x * TM >= cnt) return;
  const int m0     = start + blockIdx.x * TM;
  const int rowEnd = start + cnt;
  const int n0     = blockIdx.y * TN;

  const int tid  = threadIdx.x;
  const int lane = tid & 31;
  const int wid  = tid >> 5;
  const int wm   = wid >> 1;
  const int wn   = wid & 1;
  const int sel  = lane >> 4;
  const int l15  = lane & 15;

  v8f acc[2][2] = {};

  const float* w2b = w2 + ((size_t)e * DIM_IN + n0) * DIM_HID;

  for (int k0 = 0; k0 < DIM_HID; k0 += TK) {
    // ---- A tile: 128 x 64 bf16 copied from hbuf (async direct-to-LDS)
    #pragma unroll
    for (int p = 0; p < 4; ++p) {
      int idx  = p * 256 + tid;
      int r    = idx >> 3;               // 8 groups of 8 bf16 per 64-col row
      int c8   = (idx & 7) * 8;
      int grow = m0 + r;
      __bf16* ldst = &sA[r * LDSB + c8];
      if (grow < rowEnd) {
#if HAVE_ASYNC_LDS
        __builtin_amdgcn_global_load_async_to_lds_b128(
            (g_v4i*)(hbuf + (size_t)grow * DIM_HID + k0 + c8),
            (l_v4i*)ldst,
            0, 0);
#else
        *(uint4*)ldst = *(const uint4*)(hbuf + (size_t)grow * DIM_HID + k0 + c8);
#endif
      } else {
        *(uint4*)ldst = make_uint4(0u, 0u, 0u, 0u);   // zero-pad tail rows
      }
    }
    // ---- B tile: 64 x 64 f32 from w2 -> bf16 (4 passes x float4)
    #pragma unroll
    for (int p = 0; p < 4; ++p) {
      int idx = p * 256 + tid;
      int r = idx >> 4;
      int c = (idx & 15) * 4;
      float4 v = *(const float4*)(w2b + (size_t)r * DIM_HID + k0 + c);
      *(uint2*)&sB[r * LDSB + c] = pack4bf(v.x, v.y, v.z, v.w);
    }
    wait_async0();                       // drain ASYNCcnt before the barrier
    __syncthreads();

    #pragma unroll
    for (int ks = 0; ks < 2; ++ks) {
      const int ko = ks * 32;
      v16bf a[2], bfr[2];
      #pragma unroll
      for (int i = 0; i < 2; ++i)
        a[i] = load_fragA(&sA[(wm * 32 + i * 16 + l15) * LDSB + ko], sel);
      #pragma unroll
      for (int j = 0; j < 2; ++j)
        bfr[j] = load_fragB(&sB[(wn * 32 + j * 16 + l15) * LDSB + ko], sel);
      #pragma unroll
      for (int i = 0; i < 2; ++i)
        #pragma unroll
        for (int j = 0; j < 2; ++j)
          acc[i][j] = __builtin_amdgcn_wmma_f32_16x16x32_bf16(
              false, a[i], false, bfr[j], (short)0, acc[i][j], false, false);
    }
    __syncthreads();
  }

  // ---- epilogue: apply per-row RMS scale on f32 accumulators, store f32
  #pragma unroll
  for (int i = 0; i < 2; ++i) {
    #pragma unroll
    for (int j = 0; j < 2; ++j) {
      const int n = n0 + wn * 32 + j * 16 + l15;
      #pragma unroll
      for (int v = 0; v < 8; ++v) {
        const int m    = wm * 32 + i * 16 + v + sel * 8;
        const int grow = m0 + m;
        if (grow < rowEnd)
          out[(size_t)grow * DIM_IN + n] = scale[grow] * acc[i][j][v];
      }
    }
  }
}

// ============================================================================
// Launch. Workspace layout: hbuf bf16 [TOKENS*DIM_HID] (8 MB) + scale f32
// [TOKENS] (16 KB) = ~8.4 MB total.
// Input order per setup_inputs(): x, w1, w2, w3, mid_w, num_tokens_per_expert.
// ============================================================================
extern "C" void kernel_launch(void* const* d_in, const int* in_sizes, int n_in,
                              void* d_out, int out_size, void* d_ws, size_t ws_size,
                              hipStream_t stream) {
  const float* x      = (const float*)d_in[0];
  const float* w1     = (const float*)d_in[1];
  const float* w2     = (const float*)d_in[2];
  const float* w3     = (const float*)d_in[3];
  const float* midw   = (const float*)d_in[4];
  const int*   counts = (const int*)d_in[5];
  float*       out    = (float*)d_out;

  unsigned short* hbuf  = (unsigned short*)d_ws;
  float*          scale = (float*)((char*)d_ws + (size_t)TOKENS * DIM_HID * sizeof(unsigned short));

  dim3 blk(256, 1, 1);

  moe_up_kernel<<<dim3(MTILES, DIM_HID / TN, NUM_EXPERTS), blk, 0, stream>>>(
      x, w1, w3, midw, counts, hbuf);

  rms_scale_kernel<<<dim3(TOKENS / 8, 1, 1), blk, 0, stream>>>(hbuf, scale);

  moe_down_kernel<<<dim3(MTILES, DIM_IN / TN, NUM_EXPERTS), blk, 0, stream>>>(
      hbuf, scale, w2, counts, out);
}
